// TA_block_31619549233987
// MI455X (gfx1250) — matrix operand order repaired and checked by reading, not compiled, not run on hardware
//
#include <hip/hip_runtime.h>
#include <hip/hip_bf16.h>
#include <stdint.h>

#define DI __device__ __forceinline__

namespace {
constexpr int Nn = 4, Cn = 128, Tn = 512, Vn = 25;
constexpr int Hn = 16, On = 128, VP = 32;          // V padded to 32 for WMMA N-dim
constexpr int TV = Tn * Vn;                        // 12800
constexpr float EPS = 1e-5f;

typedef __attribute__((ext_vector_type(16))) _Float16 v16h;
typedef __attribute__((ext_vector_type(8)))  float    v8f;

union ABf16 { v16h v; uint32_t u[8]; };

DI uint32_t pack2h(float a, float b) {
  union { _Float16 h[2]; uint32_t u; } p;
  p.h[0] = (_Float16)a; p.h[1] = (_Float16)b;
  return p.u;
}

// tanh(x) = 1 - 2/(exp(2x)+1); exp via v_exp_f32, divide via v_rcp_f32.
DI float tanh_fast(float x) {
  float e = __builtin_amdgcn_exp2f(x * 2.8853900817779268f); // 2*log2(e)
  return 1.0f - 2.0f * __builtin_amdgcn_rcpf(e + 1.0f);
}

DI v8f wmma_f16(const ABf16& a, const ABf16& b, v8f c) {
  // (neg_a, A, neg_b, B, c_mod, C, reuse_a, reuse_b)
  return __builtin_amdgcn_wmma_f32_16x16x32_f16(false, a.v, false, b.v,
                                                (short)0, c, false, false);
}
} // namespace

// ---------------------------------------------------------------- K0: zero BN accumulators
__global__ void k_zero(float* __restrict__ p, int n) {
  int i = blockIdx.x * blockDim.x + threadIdx.x;
  if (i < n) p[i] = 0.f;
}

// ---------------------------------------------------------------- K1: xm (mean over V) + q,k projections
__global__ void k_meanqk(const float* __restrict__ x,
                         const float* __restrict__ Wq, const float* __restrict__ bq,
                         const float* __restrict__ Wk, const float* __restrict__ bk,
                         float* __restrict__ q, float* __restrict__ k) {
  int n = blockIdx.x / Tn, t = blockIdx.x % Tn;
  __shared__ float xm[Cn];
  int c = threadIdx.x;
  const float* xp = x + (((size_t)n * Cn + c) * Tn + t) * Vn;
  float s = 0.f;
#pragma unroll
  for (int v = 0; v < Vn; ++v) s += xp[v];
  xm[c] = s * (1.0f / Vn);
  __syncthreads();
  if (threadIdx.x < 2 * Hn) {
    int h = threadIdx.x & (Hn - 1);
    bool isK = threadIdx.x >= Hn;
    const float* W = isK ? Wk : Wq;
    float acc = isK ? bk[h] : bq[h];
#pragma unroll 8
    for (int cc = 0; cc < Cn; ++cc) acc += W[h * Cn + cc] * xm[cc];
    (isK ? k : q)[((size_t)n * Hn + h) * Tn + t] = acc;
  }
}

// ---------------------------------------------------------------- K2: v = Wv @ x + bv  (WMMA GEMM, f16 out, transposed [n][o][v'][t])
__global__ __launch_bounds__(256) void k_vproj(const float* __restrict__ x,
                                               const float* __restrict__ Wv,
                                               const float* __restrict__ bv,
                                               _Float16* __restrict__ vws) {
  int n   = blockIdx.x / (TV / 16);
  int tv0 = (blockIdx.x % (TV / 16)) * 16;
  int lane = threadIdx.x & 31;
  int ot   = threadIdx.x >> 5;              // o-tile 0..7 (one per wave)
  int mrow = lane & 15;
  int kb   = (lane < 16) ? 0 : 8;           // A-operand K sub-offset
  int kbB  = (lane < 16) ? 0 : 16;          // unused (B operand uses kb, same k order as A)

  // A tiles: Wv[16o x 32c] per K-chunk (4 chunks cover C=128)
  ABf16 A[4];
#pragma unroll
  for (int kc = 0; kc < 4; ++kc)
#pragma unroll
    for (int r = 0; r < 8; ++r) {
      int c = kc * 32 + (r & 3) * 2 + ((r >= 4) ? 16 : 0) + kb;
      const float* wp = Wv + (size_t)(ot * 16 + mrow) * Cn + c;
      A[kc].u[r] = pack2h(wp[0], wp[1]);
    }

  v8f d = {};
  int tv = tv0 + mrow;                      // B/D column (t*V+v flattened)
#pragma unroll
  for (int kc = 0; kc < 4; ++kc) {
    ABf16 B;
#pragma unroll
    for (int i = 0; i < 8; ++i) {
      int c = kc * 32 + (i & 3) * 2 + ((i >= 4) ? 16 : 0) + kb;
      const float* xp = x + ((size_t)n * Cn + c) * TV + tv;
      B.u[i] = pack2h(xp[0], xp[TV]);       // (c, c+1) for this column
    }
    d = wmma_f16(A[kc], B, d);
  }

  int t = tv / Vn, vv = tv % Vn;
  int ob = ot * 16 + ((lane < 16) ? 0 : 8);
#pragma unroll
  for (int r = 0; r < 8; ++r) {
    int o = ob + r;
    vws[(((size_t)n * On + o) * VP + vv) * Tn + t] = (_Float16)(d[r] + bv[o]);
  }
}

// ---------------------------------------------------------------- K3: fused tanh-attn + Wr combine + attn@v  (all WMMA, attn lives in LDS)
__global__ __launch_bounds__(512) void k_attn(const float* __restrict__ q,
                                              const float* __restrict__ k,
                                              const float* __restrict__ Wr,
                                              const float* __restrict__ br,
                                              const _Float16* __restrict__ vws,
                                              float* __restrict__ yws,
                                              float* __restrict__ sums,
                                              float* __restrict__ sumsq) {
  extern __shared__ char smem[];
  _Float16* attn = (_Float16*)smem;                    // [128o][16g][32t]  f16 = 128KB
  _Float16* th   = (_Float16*)(smem + 131072);         // [512 gt][16h]     f16 = 16KB
  float*    qs   = (float*)(smem + 131072 + 16384);    // [16h][16g]
  float*    brs  = qs + Hn * 16;                       // [128]

  int n  = blockIdx.x >> 5;
  int g0 = (blockIdx.x & 31) << 4;
  int tid = threadIdx.x;
  int lane = tid & 31;
  int w = tid >> 5;                                    // wave 0..15
  int mrow = lane & 15;
  int kb = (lane < 16) ? 0 : 8;

  if (tid < Hn * 16) {
    int h = tid >> 4, g = tid & 15;
    qs[tid] = q[((size_t)n * Hn + h) * Tn + g0 + g];
  }
  if (tid < On) brs[tid] = br[tid];

  // Preload Wr A tiles once: [16o x 32K], K = h (16 real, 16 zero-padded)
  ABf16 wrA[8];
#pragma unroll
  for (int ot = 0; ot < 8; ++ot)
#pragma unroll
    for (int r = 0; r < 8; ++r) {
      uint32_t pk = 0u;
      if (r < 4) {
        int h = r * 2 + kb;
        const float* wp = Wr + (size_t)(ot * 16 + mrow) * Hn + h;
        pk = pack2h(wp[0], wp[1]);
      }
      wrA[ot].u[r] = pk;
    }

  v8f acc[8][2];
#pragma unroll
  for (int oo = 0; oo < 8; ++oo) { acc[oo][0] = {}; acc[oo][1] = {}; }

  __syncthreads();                                     // qs, brs ready

  int th_h = tid & 15;
  int th_t = (tid >> 4) & 31;

  for (int t0 = 0; t0 < Tn; t0 += 32) {
    // ---- tanh tile (computed once per block, shared by all 16 waves)
    float kv = k[((size_t)n * Hn + th_h) * Tn + t0 + th_t];
#pragma unroll
    for (int g = 0; g < 16; ++g)
      th[(g * 32 + th_t) * 16 + th_h] = (_Float16)tanh_fast(qs[th_h * 16 + g] - kv);
    __syncthreads();  // th ready AND previous stage-2 done reading attn

    // ---- combine: attn[o,gt] = Wr(o,h) * tanh(h,gt) + br  (wave w -> gt tiles 2w, 2w+1)
#pragma unroll
    for (int jj = 0; jj < 2; ++jj) {
      int col = (w * 2 + jj) * 16 + (lane & 15);       // gt column = g*32 + t
      ABf16 B;
#pragma unroll
      for (int i = 0; i < 8; ++i)                      // K>=16 (elements 8..15) is zero pad
        B.u[i] = (i < 4) ? *(const uint32_t*)&th[col * 16 + 2 * i + kb] : 0u;
      int g = col >> 5, tl = col & 31;
      int obb = (lane < 16) ? 0 : 8;
#pragma unroll
      for (int ot = 0; ot < 8; ++ot) {
        v8f z = {};
        v8f d = wmma_f16(wrA[ot], B, z);
#pragma unroll
        for (int r = 0; r < 8; ++r) {
          int o = ot * 16 + obb + r;
          attn[(o * 16 + g) * 32 + tl] = (_Float16)(d[r] + brs[o]);
        }
      }
    }
    __syncthreads();  // attn ready

    // ---- stage 2: y[o,g,v'] += attn[o,g,t(32)] @ v[o,t(32),v'] (wave w -> o = 8w..8w+7)
#pragma unroll
    for (int oo = 0; oo < 8; ++oo) {
      int o = w * 8 + oo;
      ABf16 A;
#pragma unroll
      for (int r = 0; r < 8; ++r) {
        int tl = 2 * r + ((lane < 16) ? 0 : 16);
        A.u[r] = *(const uint32_t*)&attn[(o * 16 + mrow) * 32 + tl];
      }
#pragma unroll
      for (int vt = 0; vt < 2; ++vt) {
        ABf16 B;
        const _Float16* vp = vws + (((size_t)n * On + o) * VP + vt * 16 + (lane & 15)) * Tn
                             + t0 + ((lane < 16) ? 0 : 16);
#pragma unroll
        for (int i = 0; i < 8; ++i) B.u[i] = *(const uint32_t*)&vp[2 * i];
        acc[oo][vt] = wmma_f16(A, B, acc[oo][vt]);
      }
    }
    // loop-top barrier covers attn-reuse hazard; stage-2 never touches th
  }

  // ---- epilogue: store y (mask padded v') + per-channel BN partial sums
  int gb = g0 + ((lane < 16) ? 0 : 8);
#pragma unroll
  for (int oo = 0; oo < 8; ++oo) {
    int o = w * 8 + oo;
    float s1 = 0.f, s2 = 0.f;
#pragma unroll
    for (int vt = 0; vt < 2; ++vt) {
      int vv = vt * 16 + (lane & 15);
      if (vv < Vn) {
#pragma unroll
        for (int r = 0; r < 8; ++r) {
          float val = acc[oo][vt][r];
          yws[(((size_t)n * On + o) * Tn + gb + r) * Vn + vv] = val;
          s1 += val;
          s2 += val * val;
        }
      }
    }
#pragma unroll
    for (int off = 16; off > 0; off >>= 1) {
      s1 += __shfl_xor(s1, off, 32);
      s2 += __shfl_xor(s2, off, 32);
    }
    if (lane == 0) {
      atomicAdd(&sums[o], s1);
      atomicAdd(&sumsq[o], s2);
    }
  }
}

// ---------------------------------------------------------------- K4: fold BN stats into per-channel scale/shift
__global__ void k_stats(const float* __restrict__ sums, const float* __restrict__ sumsq,
                        const float* __restrict__ gamma, const float* __restrict__ beta,
                        float* __restrict__ bnsc, float* __restrict__ bnsh) {
  int o = threadIdx.x;
  if (o < On) {
    float inv = 1.0f / (float)(Nn * Tn * Vn);
    float mu  = sums[o] * inv;
    float var = sumsq[o] * inv - mu * mu;
    float sc  = gamma[o] * rsqrtf(var + EPS);
    bnsc[o] = sc;
    bnsh[o] = beta[o] - mu * sc;
  }
}

// ---------------------------------------------------------------- K5: BN + residual + ReLU
__global__ void k_final(const float* __restrict__ yws, const float* __restrict__ x,
                        const float* __restrict__ bnsc, const float* __restrict__ bnsh,
                        float* __restrict__ out) {
  size_t idx = (size_t)blockIdx.x * blockDim.x + threadIdx.x;
  if (idx >= (size_t)Nn * Cn * Tn * Vn) return;
  int o = (int)((idx / TV) % Cn);
  float val = yws[idx] * bnsc[o] + bnsh[o] + x[idx];
  out[idx] = fmaxf(val, 0.f);
}

// ---------------------------------------------------------------- launch
extern "C" void kernel_launch(void* const* d_in, const int* in_sizes, int n_in,
                              void* d_out, int out_size, void* d_ws, size_t ws_size,
                              hipStream_t stream) {
  (void)in_sizes; (void)n_in; (void)out_size; (void)ws_size;
  const float* x     = (const float*)d_in[0];
  const float* Wq    = (const float*)d_in[1];
  const float* bq    = (const float*)d_in[2];
  const float* Wk    = (const float*)d_in[3];
  const float* bk    = (const float*)d_in[4];
  const float* Wv    = (const float*)d_in[5];
  const float* bv    = (const float*)d_in[6];
  const float* Wr    = (const float*)d_in[7];
  const float* br    = (const float*)d_in[8];
  const float* gamma = (const float*)d_in[9];
  const float* beta  = (const float*)d_in[10];
  float* out = (float*)d_out;

  // workspace layout (~41.5 MB total)
  char* ws = (char*)d_ws;
  float*    qb    = (float*)(ws + 0);                       // 131072 B
  float*    kb    = (float*)(ws + 131072);                  // 131072 B
  float*    sums  = (float*)(ws + 262144);                  // 512 B
  float*    sumsq = (float*)(ws + 262656);                  // 512 B
  float*    bnsc  = (float*)(ws + 263168);                  // 512 B
  float*    bnsh  = (float*)(ws + 263680);                  // 512 B
  _Float16* vws   = (_Float16*)(ws + 524288);               // 16 MB
  float*    yws   = (float*)(ws + 524288 + 16777216);       // 25 MB

  constexpr unsigned SMEM = 131072u + 16384u + 1024u + 512u; // 148992 B (<320 KB WGP LDS)
  // opt in to >default dynamic LDS (no-op under graph replay; idempotent)
  (void)hipFuncSetAttribute(reinterpret_cast<const void*>(k_attn),
                            hipFuncAttributeMaxDynamicSharedMemorySize, (int)SMEM);

  k_zero  <<<1, 256, 0, stream>>>(sums, 256);               // sums+sumsq contiguous
  k_meanqk<<<Nn * Tn, Cn, 0, stream>>>(x, Wq, bq, Wk, bk, qb, kb);
  k_vproj <<<Nn * (TV / 16), 256, 0, stream>>>(x, Wv, bv, vws);
  k_attn  <<<Nn * (Tn / 16), 512, SMEM, stream>>>(qb, kb, Wr, br, vws, yws, sums, sumsq);
  k_stats <<<1, 128, 0, stream>>>(sums, sumsq, gamma, beta, bnsc, bnsh);
  size_t total = (size_t)Nn * Cn * Tn * Vn;
  k_final <<<(unsigned)((total + 255) / 256), 256, 0, stream>>>(yws, x, bnsc, bnsh, out);
}
